// MolecularVQVAE_16939351016122
// MI455X (gfx1250) — matrix operand order repaired and checked by reading, not compiled, not run on hardware
//
#include <hip/hip_runtime.h>
#include <hip/hip_bf16.h>
#include <stdint.h>

typedef __bf16 bf16_t;
typedef bf16_t v16bf __attribute__((ext_vector_type(16)));
typedef float  v8f   __attribute__((ext_vector_type(8)));
typedef float  v2f   __attribute__((ext_vector_type(2)));

#define THREADS 256
#define ROWS    128     // rows per workgroup
#define HDIM    256
#define DLAT    64
#define KCB     512
#define AST     264     // bf16 activation row stride (264*2 B = 16-byte aligned rows)
#define ZST     68      // f32 latent row stride (68*4 B = 16-byte aligned rows)

// ---- dynamic LDS layout (bytes) ----
#define OFF_W     0                                  // 256*256 bf16 = 131072 B (aliased: 512*64 f32 codebook)
#define OFF_ACTA  131072
#define OFF_ACTB  (OFF_ACTA + ROWS*AST*2)            // 198656
#define OFF_Z     (OFF_ACTB + ROWS*AST*2)            // 266240 (also scratch for enc_w0/b0 staging)
#define OFF_ENORM (OFF_Z + ROWS*ZST*4)               // 301056
#define OFF_IDX   (OFF_ENORM + KCB*4)                // 303104
#define OFF_RED   (OFF_IDX + ROWS*4)                 // 303616
#define OFF_X     (OFF_RED + 16*4)                   // 303680
#define OFF_P     (OFF_X + ROWS*3*4)                 // 305216: staged bias/gamma/beta
#define SMEM_BYTES (OFF_P + 3*HDIM*4)                // 308288 < 320KB WGP LDS

union BFrag { uint4 q[2]; v16bf v; };                // 32 B
union H8    { uint4 q; bf16_t h[8]; };               // 16 B
union H4    { uint2 u; bf16_t h[4]; };               // 8 B

// ------------------------------------------------------------------
// bf16 WMMA GEMM with fused bias + ReLU + LayerNorm epilogue.
//   out[128][NDIM] = LN(relu(src[128][KDIM] @ W[NDIM][KDIM]^T + bias))
// Each wave computes a 16-row stripe; NDIM/16 tiles of 16 cols each.
// A frag (16x32 bf16): lane M=l%16; VGPR v: K = 2*(v&3)+(v>>2)*16+half*8
//   -> per lane two contiguous 16B LDS chunks.
// B frag (32x16 bf16): lane N=l%16; VGPR v: K = 2*(v&3)+(v>>2)*16+half*8
//   -> per lane two contiguous 16B chunks, as for A (two b128 loads). Two base
//      pointers (tiles 0-7 / 8-15) keep all DS offsets in immediates.
// C/D (16x16 f32): lane holds rows M=j+8*half, col N=l%16; the 16 lanes
//   of a half-wave jointly hold complete rows -> LN stats via shfl_xor.
// ------------------------------------------------------------------
template<int KDIM, int NDIM, bool F32OUT>
__device__ __forceinline__ void wmma_gemm(const bf16_t* __restrict__ src, int sst,
                                          const bf16_t* __restrict__ W,
                                          const float*  __restrict__ bias,   // LDS, [NDIM]
                                          const float*  __restrict__ lng,    // LDS, [NDIM] (F32OUT: unused)
                                          const float*  __restrict__ lnb,    // LDS, [NDIM] (F32OUT: unused)
                                          void* dstv, int dst_st,
                                          int lane, int wave)
{
    const int R    = wave * 16;
    const int half = lane >> 4;
    const int l16  = lane & 15;
    const v8f vzero = {0.f,0.f,0.f,0.f,0.f,0.f,0.f,0.f};
    v8f acc[NDIM/16];
#pragma unroll
    for (int t = 0; t < NDIM/16; ++t) acc[t] = vzero;

    const bf16_t* Wl = &W[l16*KDIM];                 // tiles 0..7  base (per-lane row)
    const bf16_t* Wh = Wl + 8*16*KDIM;               // tiles 8..15 base

    for (int kt = 0; kt < KDIM/32; ++kt) {
        BFrag a;
        const bf16_t* ap = &src[(R + l16)*sst + kt*32 + half*8];
        a.q[0] = *(const uint4*)(ap);                // K = base+0..7
        a.q[1] = *(const uint4*)(ap + 16);           // K = base+16..23
#pragma unroll
        for (int t = 0; t < NDIM/16; ++t) {
            BFrag b;
            const bf16_t* bp = ((t < 8) ? Wl + t*16*KDIM : Wh + (t-8)*16*KDIM)
                               + kt*32 + half*8;
            b.q[0] = *(const uint4*)(bp);            // K = base+0..7
            b.q[1] = *(const uint4*)(bp + 16);       // K = base+16..23
            acc[t] = __builtin_amdgcn_wmma_f32_16x16x32_bf16(
                         false, a.v, false, b.v, (short)0, acc[t], false, false);
        }
    }

    if constexpr (F32OUT) {
#pragma unroll
        for (int t = 0; t < NDIM/16; ++t)
#pragma unroll
            for (int j = 0; j < 8; ++j) {
                const int r = R + j + half*8;
                const int c = t*16 + l16;
                ((float*)dstv)[r*dst_st + c] = acc[t][j] + bias[c];
            }
    } else {
        // fused bias + ReLU + per-row LayerNorm (in registers + shfl)
        float p[8], p2[8];
#pragma unroll
        for (int j = 0; j < 8; ++j) { p[j] = 0.f; p2[j] = 0.f; }
#pragma unroll
        for (int t = 0; t < NDIM/16; ++t)
#pragma unroll
            for (int j = 0; j < 8; ++j) {
                float v = acc[t][j] + bias[t*16 + l16];
                v = v > 0.f ? v : 0.f;
                acc[t][j] = v;
                p[j] += v;
                p2[j] = fmaf(v, v, p2[j]);
            }
#pragma unroll
        for (int m = 1; m < 16; m <<= 1)
#pragma unroll
            for (int j = 0; j < 8; ++j) {
                p[j]  += __shfl_xor(p[j],  m);
                p2[j] += __shfl_xor(p2[j], m);
            }
        float mu[8], rs[8];
#pragma unroll
        for (int j = 0; j < 8; ++j) {
            mu[j] = p[j] * (1.f/(float)NDIM);
            const float var = p2[j] * (1.f/(float)NDIM) - mu[j]*mu[j];
            rs[j] = rsqrtf(var + 1e-5f);
        }
        bf16_t* dst = (bf16_t*)dstv;
#pragma unroll
        for (int t = 0; t < NDIM/16; ++t)
#pragma unroll
            for (int j = 0; j < 8; ++j) {
                const int r = R + j + half*8;
                const int c = t*16 + l16;
                const float v = (acc[t][j] - mu[j]) * rs[j] * lng[c] + lnb[c];
                dst[r*dst_st + c] = (bf16_t)v;
            }
    }
}

// standalone LayerNorm (only for the non-WMMA encoder layer 0)
__device__ __forceinline__ void layernorm_128(bf16_t* act, int st,
                                              const float* __restrict__ g,
                                              const float* __restrict__ b, int tid)
{
    const int row = tid >> 1;
    const int ch  = (tid & 1) * 128;
    bf16_t* base  = &act[row*st + ch];
    float s = 0.f, s2 = 0.f;
    for (int c = 0; c < 128; c += 8) {
        H8 p; p.q = *(const uint4*)(base + c);
#pragma unroll
        for (int j = 0; j < 8; ++j) { float v = (float)p.h[j]; s += v; s2 = fmaf(v, v, s2); }
    }
    s  += __shfl_xor(s, 1);
    s2 += __shfl_xor(s2, 1);
    const float mu   = s * (1.f/256.f);
    const float var  = s2 * (1.f/256.f) - mu*mu;
    const float rstd = rsqrtf(var + 1e-5f);
    for (int c = 0; c < 128; c += 8) {
        H8 p; p.q = *(const uint4*)(base + c);
#pragma unroll
        for (int j = 0; j < 8; ++j) {
            float v = (float)p.h[j];
            p.h[j] = (bf16_t)((v - mu)*rstd*g[ch + c + j] + b[ch + c + j]);
        }
        *(uint4*)(base + c) = p.q;
    }
}

// stage f32 weights -> bf16 LDS, float4 global loads + 8B LDS stores
__device__ __forceinline__ void load_w_bf16(bf16_t* dst, const float* __restrict__ src,
                                            int n, int tid)
{
    const float4* s4 = (const float4*)src;
    for (int i = tid; i < n/4; i += THREADS) {
        float4 f = s4[i];
        H4 p;
        p.h[0] = (bf16_t)f.x; p.h[1] = (bf16_t)f.y;
        p.h[2] = (bf16_t)f.z; p.h[3] = (bf16_t)f.w;
        *(uint2*)&dst[i*4] = p.u;
    }
}

__device__ __forceinline__ void stage_vec(float* dst, const float* __restrict__ src,
                                          int n, int tid)
{
    for (int i = tid; i < n; i += THREADS) dst[i] = src[i];
}

// ------------------------------------------------------------------
// VQ argmin via fp32 WMMA 16x16x4: G = Z(16x64) @ CB^T(64x16-tile).
// dist-equivalent per col: ||e||^2 - 2*(z.e)   (||z||^2 is row-constant)
// ------------------------------------------------------------------
__device__ __forceinline__ void vq_argmin(const float* __restrict__ z,
                                          const float* __restrict__ cb,
                                          const float* __restrict__ enorm,
                                          int* __restrict__ idxb,
                                          int lane, int wave)
{
    const int R    = wave * 16;
    const int half = lane >> 4;
    const int l16  = lane & 15;
    const v8f vzero = {0.f,0.f,0.f,0.f,0.f,0.f,0.f,0.f};
    float bestd[8]; int besti[8];
#pragma unroll
    for (int j = 0; j < 8; ++j) { bestd[j] = 3.4e38f; besti[j] = 0; }

    for (int nt = 0; nt < KCB/16; ++nt) {
        v8f acc = vzero;
        for (int kt = 0; kt < DLAT/4; ++kt) {
            v2f a = *(const v2f*)&z [(R + l16)*ZST      + kt*4 + half*2];
            v2f b = *(const v2f*)&cb[(nt*16 + l16)*DLAT + kt*4 + half*2];
            acc = __builtin_amdgcn_wmma_f32_16x16x4_f32(
                      false, a, false, b, (short)0, acc, false, false);
        }
        const int col = nt*16 + l16;
        const float en = enorm[col];
#pragma unroll
        for (int j = 0; j < 8; ++j) {
            float d = en - 2.f*acc[j];
            if (d < bestd[j]) { bestd[j] = d; besti[j] = col; }
        }
    }
#pragma unroll
    for (int m = 1; m < 16; m <<= 1)
#pragma unroll
        for (int j = 0; j < 8; ++j) {
            float od = __shfl_xor(bestd[j], m);
            int   oi = __shfl_xor(besti[j], m);
            if (od < bestd[j]) { bestd[j] = od; besti[j] = oi; }
        }
    if (l16 == 0) {
#pragma unroll
        for (int j = 0; j < 8; ++j) idxb[R + j + half*8] = besti[j];
    }
}

// ------------------------------------------------------------------
__global__ void __launch_bounds__(THREADS)
vqvae_fused(const float* __restrict__ x,
            const float* __restrict__ enc_w0,  const float* __restrict__ enc_b0,
            const float* __restrict__ enc_wh,  const float* __restrict__ enc_bh,
            const float* __restrict__ enc_ln_g,const float* __restrict__ enc_ln_b,
            const float* __restrict__ enc_wout,const float* __restrict__ enc_bout,
            const float* __restrict__ codebook,
            const float* __restrict__ dec_w0,  const float* __restrict__ dec_b0,
            const float* __restrict__ dec_wh,  const float* __restrict__ dec_bh,
            const float* __restrict__ dec_ln_g,const float* __restrict__ dec_ln_b,
            const float* __restrict__ dec_wout,const float* __restrict__ dec_bout,
            float* __restrict__ out_recon, float* __restrict__ out_quant,
            float* __restrict__ ws)
{
    extern __shared__ char smem[];
    bf16_t* wsm   = (bf16_t*)(smem + OFF_W);
    float*  cbf   = (float*)(smem + OFF_W);        // aliases wsm during VQ
    bf16_t* actA  = (bf16_t*)(smem + OFF_ACTA);
    bf16_t* actB  = (bf16_t*)(smem + OFF_ACTB);
    float*  zbuf  = (float*)(smem + OFF_Z);
    float*  w0s   = (float*)(smem + OFF_Z);        // layer-0 weight staging (pre-zbuf use)
    float*  b0s   = w0s + HDIM*3;
    float*  enorm = (float*)(smem + OFF_ENORM);
    int*    idxb  = (int*)  (smem + OFF_IDX);
    float*  red   = (float*)(smem + OFF_RED);
    float*  xs    = (float*)(smem + OFF_X);
    float*  pbias = (float*)(smem + OFF_P);        // staged per-layer params
    float*  pg    = pbias + HDIM;
    float*  pb    = pg + HDIM;

    const int tid  = threadIdx.x;
    const int lane = tid & 31;
    const int wave = tid >> 5;
    const int row0 = blockIdx.x * ROWS;

    // stage x tile + layer-0 params
    for (int i = tid; i < ROWS*3; i += THREADS) xs[i] = x[row0*3 + i];
    for (int i = tid; i < HDIM*3; i += THREADS) w0s[i] = enc_w0[i];
    if (tid < HDIM) b0s[tid] = enc_b0[tid];
    __syncthreads();

    // ---------------- encoder layer 0 (3 -> 256, VALU, b128 stores) ----------------
    for (int t = tid; t < ROWS*(HDIM/8); t += THREADS) {
        const int r  = t >> 5;           // HDIM/8 = 32 col-groups
        const int cg = (t & 31) * 8;
        const float x0 = xs[r*3+0], x1 = xs[r*3+1], x2 = xs[r*3+2];
        H8 p;
#pragma unroll
        for (int j = 0; j < 8; ++j) {
            const int c = cg + j;
            float v = fmaf(x0, w0s[c*3+0], fmaf(x1, w0s[c*3+1], fmaf(x2, w0s[c*3+2], b0s[c])));
            p.h[j] = (bf16_t)(v > 0.f ? v : 0.f);
        }
        *(uint4*)&actA[r*AST + cg] = p.q;
    }
    __syncthreads();
    layernorm_128(actA, AST, enc_ln_g, enc_ln_b, tid);
    __syncthreads();

    // ---------------- encoder hidden 1 (fused LN) ----------------
    load_w_bf16(wsm, enc_wh, HDIM*HDIM, tid);
    stage_vec(pbias, enc_bh,            HDIM, tid);
    stage_vec(pg,    enc_ln_g + HDIM,   HDIM, tid);
    stage_vec(pb,    enc_ln_b + HDIM,   HDIM, tid);
    __syncthreads();
    wmma_gemm<HDIM, HDIM, false>(actA, AST, wsm, pbias, pg, pb, actB, AST, lane, wave);
    __syncthreads();

    // ---------------- encoder hidden 2 (fused LN) ----------------
    load_w_bf16(wsm, enc_wh + HDIM*HDIM, HDIM*HDIM, tid);
    stage_vec(pbias, enc_bh + HDIM,     HDIM, tid);
    stage_vec(pg,    enc_ln_g + 2*HDIM, HDIM, tid);
    stage_vec(pb,    enc_ln_b + 2*HDIM, HDIM, tid);
    __syncthreads();
    wmma_gemm<HDIM, HDIM, false>(actB, AST, wsm, pbias, pg, pb, actA, AST, lane, wave);
    __syncthreads();

    // ---------------- encoder out (256 -> 64, f32 latent) ----------------
    load_w_bf16(wsm, enc_wout, DLAT*HDIM, tid);
    stage_vec(pbias, enc_bout, DLAT, tid);
    __syncthreads();
    wmma_gemm<HDIM, DLAT, true>(actA, AST, wsm, pbias, nullptr, nullptr, zbuf, ZST, lane, wave);
    __syncthreads();

    // ---------------- VQ: codebook (fp32) into aliased LDS slab ----------------
    {
        const float4* cs = (const float4*)codebook;
        float4* cd = (float4*)cbf;
        for (int i = tid; i < KCB*DLAT/4; i += THREADS) cd[i] = cs[i];
    }
    __syncthreads();
    for (int k = tid; k < KCB; k += THREADS) {
        float s = 0.f;
        const float4* c4 = (const float4*)&cbf[k*DLAT];
        for (int d = 0; d < DLAT/4; ++d) {
            float4 v = c4[d];
            s = fmaf(v.x, v.x, s); s = fmaf(v.y, v.y, s);
            s = fmaf(v.z, v.z, s); s = fmaf(v.w, v.w, s);
        }
        enorm[k] = s;
    }
    __syncthreads();
    vq_argmin(zbuf, cbf, enorm, idxb, lane, wave);
    __syncthreads();

    // gather quantized (float4): write output, stage bf16 for decoder, vq loss
    float qz_sum = 0.f;
    for (int i = tid; i < ROWS*DLAT/4; i += THREADS) {
        const int r = i >> 4, d4 = (i & 15) * 4;
        float4 q  = *(const float4*)&cbf[idxb[r]*DLAT + d4];
        float4 z4 = *(const float4*)&zbuf[r*ZST + d4];
        float d0 = q.x - z4.x, d1 = q.y - z4.y, d2 = q.z - z4.z, d3 = q.w - z4.w;
        qz_sum = fmaf(d0, d0, fmaf(d1, d1, fmaf(d2, d2, fmaf(d3, d3, qz_sum))));
        *(float4*)&out_quant[(size_t)(row0 + r)*DLAT + d4] = q;
        H4 p;
        p.h[0] = (bf16_t)q.x; p.h[1] = (bf16_t)q.y;
        p.h[2] = (bf16_t)q.z; p.h[3] = (bf16_t)q.w;
        *(uint2*)&actA[r*AST + d4] = p.u;
    }
    __syncthreads();

    // ---------------- decoder layer 0 (64 -> 256, fused LN) ----------------
    load_w_bf16(wsm, dec_w0, HDIM*DLAT, tid);
    stage_vec(pbias, dec_b0,   HDIM, tid);
    stage_vec(pg,    dec_ln_g, HDIM, tid);
    stage_vec(pb,    dec_ln_b, HDIM, tid);
    __syncthreads();
    wmma_gemm<DLAT, HDIM, false>(actA, AST, wsm, pbias, pg, pb, actB, AST, lane, wave);
    __syncthreads();

    // ---------------- decoder hidden 1 (fused LN) ----------------
    load_w_bf16(wsm, dec_wh, HDIM*HDIM, tid);
    stage_vec(pbias, dec_bh,            HDIM, tid);
    stage_vec(pg,    dec_ln_g + HDIM,   HDIM, tid);
    stage_vec(pb,    dec_ln_b + HDIM,   HDIM, tid);
    __syncthreads();
    wmma_gemm<HDIM, HDIM, false>(actB, AST, wsm, pbias, pg, pb, actA, AST, lane, wave);
    __syncthreads();

    // ---------------- decoder hidden 2 (fused LN) ----------------
    load_w_bf16(wsm, dec_wh + HDIM*HDIM, HDIM*HDIM, tid);
    stage_vec(pbias, dec_bh + HDIM,     HDIM, tid);
    stage_vec(pg,    dec_ln_g + 2*HDIM, HDIM, tid);
    stage_vec(pb,    dec_ln_b + 2*HDIM, HDIM, tid);
    __syncthreads();
    wmma_gemm<HDIM, HDIM, false>(actA, AST, wsm, pbias, pg, pb, actB, AST, lane, wave);
    __syncthreads();

    // ---------------- decoder out (256 -> 3) + recon loss (vectorized) ----------------
    float rec_sum = 0.f;
    for (int i = tid; i < ROWS*3; i += THREADS) {
        const int r = i / 3, c = i % 3;
        float s = dec_bout[c];
        const bf16_t* arow = &actB[r*AST];
        const float4* w4   = (const float4*)&dec_wout[c*HDIM];
        for (int k = 0; k < HDIM; k += 8) {
            H8 p; p.q = *(const uint4*)(arow + k);
            float4 wa = w4[k/4], wb = w4[k/4 + 1];
            s = fmaf((float)p.h[0], wa.x, s); s = fmaf((float)p.h[1], wa.y, s);
            s = fmaf((float)p.h[2], wa.z, s); s = fmaf((float)p.h[3], wa.w, s);
            s = fmaf((float)p.h[4], wb.x, s); s = fmaf((float)p.h[5], wb.y, s);
            s = fmaf((float)p.h[6], wb.z, s); s = fmaf((float)p.h[7], wb.w, s);
        }
        out_recon[(size_t)(row0 + r)*3 + c] = s;
        const float dx = s - xs[r*3 + c];
        rec_sum = fmaf(dx, dx, rec_sum);
    }

    // ---------------- loss reduction ----------------
    for (int m = 1; m < 32; m <<= 1) {
        qz_sum  += __shfl_xor(qz_sum,  m);
        rec_sum += __shfl_xor(rec_sum, m);
    }
    if (lane == 0) { red[wave] = qz_sum; red[8 + wave] = rec_sum; }
    __syncthreads();
    if (tid == 0) {
        float a = 0.f, b = 0.f;
        for (int i = 0; i < 8; ++i) { a += red[i]; b += red[8 + i]; }
        atomicAdd(&ws[0], a);
        atomicAdd(&ws[1], b);
    }
}

__global__ void vqvae_init(float* ws) { if (threadIdx.x < 2) ws[threadIdx.x] = 0.f; }

__global__ void vqvae_finalize(const float* __restrict__ ws, float* __restrict__ out_loss)
{
    if (threadIdx.x == 0) {
        const float nrec = 2048.f * 128.f * 3.f;
        const float nqz  = 2048.f * 128.f * 64.f;
        // forward value: commitment*e_latent + q_latent == 1.25 * mean((q-z)^2)
        *out_loss = ws[1] / nrec + 1.25f * ws[0] / nqz;
    }
}

extern "C" void kernel_launch(void* const* d_in, const int* in_sizes, int n_in,
                              void* d_out, int out_size, void* d_ws, size_t ws_size,
                              hipStream_t stream)
{
    const float* x        = (const float*)d_in[0];
    const float* enc_w0   = (const float*)d_in[1];
    const float* enc_b0   = (const float*)d_in[2];
    const float* enc_wh   = (const float*)d_in[3];
    const float* enc_bh   = (const float*)d_in[4];
    const float* enc_ln_g = (const float*)d_in[5];
    const float* enc_ln_b = (const float*)d_in[6];
    const float* enc_wout = (const float*)d_in[7];
    const float* enc_bout = (const float*)d_in[8];
    const float* codebook = (const float*)d_in[9];
    const float* dec_w0   = (const float*)d_in[10];
    const float* dec_b0   = (const float*)d_in[11];
    const float* dec_wh   = (const float*)d_in[12];
    const float* dec_bh   = (const float*)d_in[13];
    const float* dec_ln_g = (const float*)d_in[14];
    const float* dec_ln_b = (const float*)d_in[15];
    const float* dec_wout = (const float*)d_in[16];
    const float* dec_bout = (const float*)d_in[17];

    float* out_recon = (float*)d_out;                          // 2048*128*3
    float* out_quant = out_recon + (size_t)2048*128*3;         // 2048*128*64
    float* out_loss  = out_quant + (size_t)2048*128*64;        // 1
    float* ws        = (float*)d_ws;

    (void)hipFuncSetAttribute((const void*)vqvae_fused,
                              hipFuncAttributeMaxDynamicSharedMemorySize, SMEM_BYTES);

    vqvae_init<<<1, 32, 0, stream>>>(ws);
    vqvae_fused<<<2048, THREADS, SMEM_BYTES, stream>>>(
        x, enc_w0, enc_b0, enc_wh, enc_bh, enc_ln_g, enc_ln_b, enc_wout, enc_bout,
        codebook, dec_w0, dec_b0, dec_wh, dec_bh, dec_ln_g, dec_ln_b, dec_wout, dec_bout,
        out_recon, out_quant, ws);
    vqvae_finalize<<<1, 32, 0, stream>>>(ws, out_loss);
}
